// CRPSKumaraswamyLoss_77068893159438
// MI455X (gfx1250) — compile-verified, matrix operand order checked by reading, not run on hardware
//
#include <hip/hip_runtime.h>
#include <hip/hip_bf16.h>
#include <stdint.h>

// ---------------------------------------------------------------------------
// CRPS of a Kumaraswamy mixture, CDNA5 (gfx1250, wave32).
//  - Threefry-2x32-20 counter-based RNG keyed on (cell, sample): stateless,
//    deterministic, zero memory traffic for RNG state.
//  - Kumaraswamy inverse CDF via v_log_f32 / v_exp_f32 (amdgcn builtins).
//  - Sample/pair abs-diff reductions done on the matrix pipe:
//    V_WMMA_F32_16X16X32_BF16 with an all-ones B matrix => row sums,
//    f32 accumulation in C across the whole per-wave loop. Since only the
//    grand total is needed and B==1, sum over all D lanes/components equals
//    16 * (sum of all A entries) -- A layout is irrelevant.
//  - float->bf16 via native conversions (compiler packs v_cvt_*bf16*),
//    not software bit-twiddling.
//  - Samples staged in LDS (wave-private slab); wave-local ordering via
//    CDNA5 split counter: s_wait_dscnt 0.
// ---------------------------------------------------------------------------

typedef __attribute__((ext_vector_type(16))) __bf16          v16bf;
typedef __attribute__((ext_vector_type(8)))  float           v8f;
typedef __attribute__((ext_vector_type(16))) unsigned short  v16us;

#define N_CELLS         (4*4*128*128)       /* 262144 */
#define SAMPLES         100
#define PAIRS           96
#define SMP_PAD         105                 /* odd-ish pitch: conflict-free */
#define BLOCKS          512
#define WAVES_PER_BLOCK 4
#define THREADS         (WAVES_PER_BLOCK*32)
#define TOTAL_WAVES     (BLOCKS*WAVES_PER_BLOCK)   /* 2048 */
#define NGROUPS         (N_CELLS/16)               /* 16384 */
#define GROUPS_PER_WAVE (NGROUPS/TOTAL_WAVES)      /* 8 */

// Hardware transcendentals (base-2): v_log_f32 / v_exp_f32
__device__ __forceinline__ float hw_log2(float x) { return __builtin_amdgcn_logf(x); }
__device__ __forceinline__ float hw_exp2(float x) { return __builtin_amdgcn_exp2f(x); }

// Threefry-2x32, 20 rounds (JAX's PRNG family), key = (0, 42)
__device__ __forceinline__ void threefry2x32(uint32_t c0, uint32_t c1,
                                             uint32_t& o0, uint32_t& o1) {
    const uint32_t k0 = 0u, k1 = 42u;
    const uint32_t k2 = 0x1BD11BDAu ^ k0 ^ k1;
    uint32_t x0 = c0 + k0, x1 = c1 + k1;
#define TF_R(r) { x0 += x1; x1 = (x1 << (r)) | (x1 >> (32 - (r))); x1 ^= x0; }
    TF_R(13) TF_R(15) TF_R(26) TF_R(6)  x0 += k1; x1 += k2 + 1u;
    TF_R(17) TF_R(29) TF_R(16) TF_R(24) x0 += k2; x1 += k0 + 2u;
    TF_R(13) TF_R(15) TF_R(26) TF_R(6)  x0 += k0; x1 += k1 + 3u;
    TF_R(17) TF_R(29) TF_R(16) TF_R(24) x0 += k1; x1 += k2 + 4u;
    TF_R(13) TF_R(15) TF_R(26) TF_R(6)  x0 += k2; x1 += k0 + 5u;
#undef TF_R
    o0 = x0; o1 = x1;
}

__global__ void __launch_bounds__(THREADS)
crps_zero_kernel(float* out) { out[0] = 0.0f; }

__global__ void __launch_bounds__(THREADS)
crps_kumaraswamy_kernel(const float* __restrict__ alpha,
                        const float* __restrict__ beta,
                        const float* __restrict__ w,
                        const float* __restrict__ y_true,
                        const int*   __restrict__ i_idx,
                        const int*   __restrict__ j_idx,
                        float*       __restrict__ out) {
    __shared__ float smp[WAVES_PER_BLOCK][16][SMP_PAD];   // ~26.9 KB
    __shared__ int   sIp[PAIRS], sJp[PAIRS];

    const int tid = threadIdx.x;
    if (tid < PAIRS) { sIp[tid] = i_idx[tid]; sJp[tid] = j_idx[tid]; }
    __syncthreads();

    const int lane = tid & 31;
    const int wv   = tid >> 5;
    const int cLoc = lane & 15;     // which of the wave's 16 cells
    const int h    = lane >> 4;     // sample-half handled by this lane
    const int gw   = blockIdx.x * WAVES_PER_BLOCK + wv;

    const float w0 = w[0], w1 = w[1], w2 = w[2];
    const float cw0 = w0, cw1 = w0 + w1, cw2 = w0 + w1 + w2;

    // all-ones bf16 B matrix: D[m,n] = sum_k A[m,k]
    v16us ones_us;
#pragma unroll
    for (int i = 0; i < 16; ++i) ones_us[i] = 0x3F80u;
    const v16bf onesV = __builtin_bit_cast(v16bf, ones_us);

    v8f c1 = {0.f,0.f,0.f,0.f,0.f,0.f,0.f,0.f};   // sum |x - y|
    v8f c2 = {0.f,0.f,0.f,0.f,0.f,0.f,0.f,0.f};   // sum |x_i - x_j|

    const float EPSf = 1e-6f;

    for (int it = 0; it < GROUPS_PER_WAVE; ++it) {
        const int group = it * TOTAL_WAVES + gw;
        const int cg    = group * 16 + cLoc;        // global cell index

        const float4 av = reinterpret_cast<const float4*>(alpha)[cg];
        const float4 bv = reinterpret_cast<const float4*>(beta)[cg];
        const float  y  = y_true[cg];

        if (it + 1 < GROUPS_PER_WAVE) {             // global_prefetch_b8
            const int cn = ((it + 1) * TOTAL_WAVES + gw) * 16 + cLoc;
            __builtin_prefetch(alpha + (size_t)cn * 4, 0, 1);
            __builtin_prefetch(beta  + (size_t)cn * 4, 0, 1);
        }

        const float ia0 = 1.0f/av.x, ia1 = 1.0f/av.y, ia2 = 1.0f/av.z, ia3 = 1.0f/av.w;
        const float ib0 = 1.0f/bv.x, ib1 = 1.0f/bv.y, ib2 = 1.0f/bv.z, ib3 = 1.0f/bv.w;

        // ------- sample generation + term1 reduction on the matrix pipe ----
#pragma unroll
        for (int b = 0; b < 4; ++b) {
            v16bf aV;
#pragma unroll
            for (int e = 0; e < 16; ++e) {
                const int s = b * 32 + h * 16 + e;  // covers 0..127, keep <100
                float v = 0.0f;
                if (s < SAMPLES) {
                    uint32_t r0, r1;
                    threefry2x32((uint32_t)cg, (uint32_t)s, r0, r1);
                    // mixture component via inverse-CDF on cumulative weights
                    const float rc = (float)(r0 >> 8) * (1.0f/16777216.0f);
                    const int comp = (rc >= cw0) + (rc >= cw1) + (rc >= cw2);
                    const float ia = comp==0 ? ia0 : comp==1 ? ia1 : comp==2 ? ia2 : ia3;
                    const float ib = comp==0 ? ib0 : comp==1 ? ib1 : comp==2 ? ib2 : ib3;
                    // u in [EPS, 1-EPS)
                    const float u01 = (float)(r1 >> 8) * (1.0f/16777216.0f);
                    const float u   = EPSf + u01 * (1.0f - 2.0f*EPSf);
                    // Kumaraswamy inverse CDF: (1 - (1-u)^(1/b))^(1/a)
                    const float t    = 1.0f - u;
                    const float p    = hw_exp2(ib * hw_log2(t));
                    float base       = fmaxf(1.0f - p, 1e-12f);
                    const float x    = hw_exp2(ia * hw_log2(base));
                    smp[wv][cLoc][s] = x;
                    v = fabsf(x - y);
                }
                aV[e] = (__bf16)v;                  // native cvt, packs 2/op
            }
            c1 = __builtin_amdgcn_wmma_f32_16x16x32_bf16(
                     false, aV, false, onesV, (short)0, c1, false, false);
        }

        // wave-local fence: partner lane (L^16) must see our LDS writes
        asm volatile("s_wait_dscnt 0" ::: "memory");

        // ------- term2: 96 fixed pairs, 3 WMMA batches ---------------------
#pragma unroll
        for (int pb = 0; pb < 3; ++pb) {
            v16bf pV;
#pragma unroll
            for (int e = 0; e < 16; ++e) {
                const int p = pb * 32 + h * 16 + e;           // 0..95
                const float d = fabsf(smp[wv][cLoc][sIp[p]] -
                                      smp[wv][cLoc][sJp[p]]);
                pV[e] = (__bf16)d;                  // native cvt
            }
            c2 = __builtin_amdgcn_wmma_f32_16x16x32_bf16(
                     false, pV, false, onesV, (short)0, c2, false, false);
        }
    }

    // Sum over all D lanes/components = 16 * (sum of everything fed into A)
    float t1 = 0.0f, t2 = 0.0f;
#pragma unroll
    for (int r = 0; r < 8; ++r) { t1 += c1[r]; t2 += c2[r]; }
    float v = t1 * (1.0f / (16.0f * (float)SAMPLES))
            - t2 * (0.5f / (16.0f * (float)PAIRS));
#pragma unroll
    for (int m = 16; m > 0; m >>= 1) v += __shfl_xor(v, m, 32);
    if (lane == 0) atomicAdd(out, v * (1.0f / (float)N_CELLS));
}

extern "C" void kernel_launch(void* const* d_in, const int* in_sizes, int n_in,
                              void* d_out, int out_size, void* d_ws, size_t ws_size,
                              hipStream_t stream) {
    const float* alpha = (const float*)d_in[0];
    const float* beta  = (const float*)d_in[1];
    const float* wts   = (const float*)d_in[2];
    const float* ytru  = (const float*)d_in[3];
    const int*   iidx  = (const int*)d_in[4];
    const int*   jidx  = (const int*)d_in[5];
    float* out = (float*)d_out;

    hipLaunchKernelGGL(crps_zero_kernel, dim3(1), dim3(THREADS), 0, stream, out);
    hipLaunchKernelGGL(crps_kumaraswamy_kernel, dim3(BLOCKS), dim3(THREADS), 0, stream,
                       alpha, beta, wts, ytru, iidx, jidx, out);
}
